// UncertaintyAwareAdaptiveAttention_27376121545122
// MI455X (gfx1250) — compile-verified, matrix operand order checked by reading, not run on hardware
//
#include <hip/hip_runtime.h>
#include <hip/hip_bf16.h>
#include <math.h>

// ---------------------------------------------------------------------------
// Types for CDNA5 WMMA (wave32)
// ---------------------------------------------------------------------------
typedef __attribute__((ext_vector_type(16))) __bf16 v16bf;
typedef __attribute__((ext_vector_type(8)))  __bf16 v8bf;
typedef __attribute__((ext_vector_type(2)))  __bf16 v2bf;
typedef __attribute__((ext_vector_type(8)))  float  v8f;

#define HDIM 1024
#define SEQ  1024
#define BSZ  4
#define NHEAD 8
#define HD   128
#define LLAY 2
#define MTOK 8192                 // LLAY*BSZ*SEQ
#define NEL  8388608ull           // MTOK*HDIM
#define ACT_NONE 0
#define ACT_GELU 1

// native conversion: lowers to v_cvt (pk) bf16 ops on gfx1250
__device__ __forceinline__ __bf16 f2bf(float f) { return (__bf16)f; }

__device__ __forceinline__ unsigned pk2(float x, float y) {
  v2bf v = { (__bf16)x, (__bf16)y };   // v_cvt_pk_bf16_f32
  return __builtin_bit_cast(unsigned, v);
}

// LDS byte offset of a generic pointer into __shared__ (flat addr[31:0] == LDS addr)
__device__ __forceinline__ unsigned lds_off(const void* p) {
  return (unsigned)(uintptr_t)p;
}

// Load one 16-lane-owned A/B fragment slice: two aligned 16-byte LDS reads.
// Layout (CDNA5 ISA 7.12.2, 16-bit A 16x32, mirrored for B 32x16):
// lanes 0-15 hold K=0..7 & 16..23, lanes 16-31 hold K=8..15 & 24..31.
__device__ __forceinline__ v16bf ld_frag(const __bf16* rowp, int khalf) {
  v8bf lo = *(const v8bf*)(rowp + khalf);
  v8bf hi = *(const v8bf*)(rowp + khalf + 16);
  return __builtin_shufflevector(lo, hi, 0,1,2,3,4,5,6,7,8,9,10,11,12,13,14,15);
}

__device__ __forceinline__ float gelu_exact(float x) {
  return 0.5f * x * (1.0f + erff(x * 0.7071067811865476f));
}

// ---------------------------------------------------------------------------
// Kernel 0: weight transpose+convert  Wt[n][k] = bf16(W[k][n])
// ---------------------------------------------------------------------------
__global__ __launch_bounds__(256)
void convertT_kernel(const float* __restrict__ W, __bf16* __restrict__ Wt,
                     int K, int N) {
  __shared__ float tile[32][33];
  const int kb = blockIdx.y * 32, nb = blockIdx.x * 32;
  const int tx = threadIdx.x & 31, ty = threadIdx.x >> 5;   // ty 0..7
#pragma unroll
  for (int i = 0; i < 32; i += 8)
    tile[ty + i][tx] = W[(size_t)(kb + ty + i) * N + nb + tx];
  __syncthreads();
#pragma unroll
  for (int i = 0; i < 32; i += 8)
    Wt[(size_t)(nb + ty + i) * K + kb + tx] = f2bf(tile[tx][ty + i]);
}

// ---------------------------------------------------------------------------
// Kernel 1: tiny uncertainty/routing MLPs (B=4) in one block
// ---------------------------------------------------------------------------
__global__ __launch_bounds__(256)
void small_mlp_kernel(const float* __restrict__ umap,
                      const float* ueW1, const float* ueb1,
                      const float* ueW2, const float* ueb2,
                      const float* ueW3, const float* ueb3,
                      const float* rW1,  const float* rb1,
                      const float* rW2,  const float* rb2,
                      const float* Us1,  const float* Ub1,
                      const float* Us2,  const float* Ub2,
                      float* __restrict__ uscale,   // B*NHEAD
                      float* __restrict__ probs,    // B*3
                      float* __restrict__ dec) {    // B
  __shared__ float gu[4];
  __shared__ float x1[256];
  __shared__ float x2[512];
  __shared__ float ue[1024];
  __shared__ float red[256];
  __shared__ float lg[3];
  const int t = threadIdx.x;

  for (int b = 0; b < 4; ++b) {
    float s = 0.f;
    for (int i = t; i < 36864; i += 256) s += umap[b * 36864 + i];
    red[t] = s; __syncthreads();
    for (int o = 128; o > 0; o >>= 1) { if (t < o) red[t] += red[t + o]; __syncthreads(); }
    if (t == 0) gu[b] = red[0] * (1.f / 36864.f);
    __syncthreads();
  }
  if (t == 0) {
    float mn = fminf(fminf(gu[0], gu[1]), fminf(gu[2], gu[3]));
    float mx = fmaxf(fmaxf(gu[0], gu[1]), fmaxf(gu[2], gu[3]));
    float inv = 1.f / (mx - mn + 1e-8f);
    for (int b = 0; b < 4; ++b) gu[b] = (gu[b] - mn) * inv;
  }
  __syncthreads();

  for (int b = 0; b < 4; ++b) {
    for (int j = t; j < 256; j += 256) x1[j] = fmaxf(0.f, gu[b] * ueW1[j] + ueb1[j]);
    __syncthreads();
    for (int j = t; j < 512; j += 256) {
      float a = ueb2[j];
      for (int i = 0; i < 256; ++i) a += x1[i] * ueW2[i * 512 + j];
      x2[j] = fmaxf(0.f, a);
    }
    __syncthreads();
    for (int j = t; j < 1024; j += 256) {
      float a = ueb3[j];
      for (int i = 0; i < 512; ++i) a += x2[i] * ueW3[i * 1024 + j];
      ue[j] = a;
    }
    __syncthreads();
    for (int j = t; j < 512; j += 256) {
      float a = rb1[j];
      for (int i = 0; i < 1024; ++i) a += ue[i] * rW1[i * 512 + j];
      x2[j] = fmaxf(0.f, a);
    }
    __syncthreads();
    if (t < 3) {
      float a = rb2[t];
      for (int i = 0; i < 512; ++i) a += x2[i] * rW2[i * 3 + t];
      lg[t] = a;
    }
    __syncthreads();
    if (t == 0) {
      float m = fmaxf(lg[0], fmaxf(lg[1], lg[2]));
      float e0 = __expf(lg[0] - m), e1 = __expf(lg[1] - m), e2 = __expf(lg[2] - m);
      float inv = 1.f / (e0 + e1 + e2);
      probs[b * 3 + 0] = e0 * inv; probs[b * 3 + 1] = e1 * inv; probs[b * 3 + 2] = e2 * inv;
      int d = 0; float bv = e0;
      if (e1 > bv) { d = 1; bv = e1; }
      if (e2 > bv) { d = 2; }
      dec[b] = (float)d;
    }
    for (int j = t; j < 256; j += 256) {
      float a = Ub1[j];
      for (int i = 0; i < 1024; ++i) a += ue[i] * Us1[i * 256 + j];
      x1[j] = fmaxf(0.f, a);
    }
    __syncthreads();
    if (t < 8) {
      float a = Ub2[t];
      for (int i = 0; i < 256; ++i) a += x1[i] * Us2[i * 8 + t];
      uscale[b * 8 + t] = 1.f / (1.f + __expf(-a));
    }
    __syncthreads();
  }
}

// ---------------------------------------------------------------------------
// Kernel 2: software-pipelined WMMA bf16 GEMM
//   C[M,N] = act(A[M,K] @ W[K,N] + bias[N])
//   A: fp32 MxK, converted in-kernel via v_cvt_pk_bf16_f32, packed b32 stores
//   Wt: bf16 NxK (pre-transposed): B tiles copied with
//       global_load_async_to_lds_b128 (ASYNCcnt)
//   Double-buffered LDS; tile k+1 loads issued before tile k WMMAs;
//   one barrier per K-step.
// ---------------------------------------------------------------------------
__global__ __launch_bounds__(256)
void gemm_bias_act(const float* __restrict__ A, const __bf16* __restrict__ Wt,
                   const float* __restrict__ bias, float* __restrict__ C,
                   int M, int N, int K, int act) {
  __shared__ __bf16 As[2][128][40];   // stride 80B: 16B-aligned fragment runs
  __shared__ __bf16 Bs[2][128][40];   // [n][k]
  const int tid  = threadIdx.x;
  const int wave = tid >> 5, lid = tid & 31, half = lid >> 4, l = lid & 15;
  const int wm = wave >> 1, wn = wave & 1;
  const int rowBase = blockIdx.y * 128;
  const int colBase = blockIdx.x * 128;
  (void)M;

  // A-tile loader lanes: 4 rows x float4 per thread
  const int akc = (tid & 7) * 4;    // k col 0..28
  const int arb = tid >> 3;         // row   0..31 (4 passes of 32)
  const float* aptr = A + (size_t)(rowBase + arb) * K + akc;

  // B-tile async chunks: 128 rows x 2 x 16B per thread
  const int bn0 = tid >> 2;         // row 0..63 (and +64)
  const int bc  = (tid & 3) * 8;    // 8 bf16 = 16B chunk
  const __bf16* bptr0 = Wt + (size_t)(colBase + bn0) * K + bc;
  const __bf16* bptr1 = Wt + (size_t)(colBase + bn0 + 64) * K + bc;
  const unsigned boff0[2] = { lds_off(&Bs[0][bn0][bc]),      lds_off(&Bs[1][bn0][bc]) };
  const unsigned boff1[2] = { lds_off(&Bs[0][bn0 + 64][bc]), lds_off(&Bs[1][bn0 + 64][bc]) };

  v8f acc[2][4];
  const v8f vz = {0.f,0.f,0.f,0.f,0.f,0.f,0.f,0.f};
  for (int f = 0; f < 2; ++f)
    for (int g = 0; g < 4; ++g) acc[f][g] = vz;

  float4 st[4];   // A-tile register staging

  auto issueB = [&](int buf) {
    asm volatile("global_load_async_to_lds_b128 %0, %1, off"
                 :: "v"(boff0[buf]), "v"(bptr0) : "memory");
    asm volatile("global_load_async_to_lds_b128 %0, %1, off"
                 :: "v"(boff1[buf]), "v"(bptr1) : "memory");
    bptr0 += 32; bptr1 += 32;
  };
  auto loadA = [&]() {
#pragma unroll
    for (int p = 0; p < 4; ++p)
      st[p] = *(const float4*)(aptr + (size_t)p * 32 * K);
    aptr += 32;
  };
  auto storeA = [&](int buf) {
#pragma unroll
    for (int p = 0; p < 4; ++p) {
      unsigned* dst = (unsigned*)&As[buf][arb + p * 32][akc];
      dst[0] = pk2(st[p].x, st[p].y);
      dst[1] = pk2(st[p].z, st[p].w);
    }
  };
  auto compute = [&](int buf) {
    v16bf a0 = ld_frag(&As[buf][wm * 32 + l][0],      half * 8);
    v16bf a1 = ld_frag(&As[buf][wm * 32 + 16 + l][0], half * 8);
    v16bf b0 = ld_frag(&Bs[buf][wn * 64 +  0 + l][0], half * 8);
    v16bf b1 = ld_frag(&Bs[buf][wn * 64 + 16 + l][0], half * 8);
    v16bf b2 = ld_frag(&Bs[buf][wn * 64 + 32 + l][0], half * 8);
    v16bf b3 = ld_frag(&Bs[buf][wn * 64 + 48 + l][0], half * 8);
    acc[0][0] = __builtin_amdgcn_wmma_f32_16x16x32_bf16(false, a0, false, b0, (short)0, acc[0][0], false, false);
    acc[1][0] = __builtin_amdgcn_wmma_f32_16x16x32_bf16(false, a1, false, b0, (short)0, acc[1][0], false, false);
    acc[0][1] = __builtin_amdgcn_wmma_f32_16x16x32_bf16(false, a0, false, b1, (short)0, acc[0][1], false, false);
    acc[1][1] = __builtin_amdgcn_wmma_f32_16x16x32_bf16(false, a1, false, b1, (short)0, acc[1][1], false, false);
    acc[0][2] = __builtin_amdgcn_wmma_f32_16x16x32_bf16(false, a0, false, b2, (short)0, acc[0][2], false, false);
    acc[1][2] = __builtin_amdgcn_wmma_f32_16x16x32_bf16(false, a1, false, b2, (short)0, acc[1][2], false, false);
    acc[0][3] = __builtin_amdgcn_wmma_f32_16x16x32_bf16(false, a0, false, b3, (short)0, acc[0][3], false, false);
    acc[1][3] = __builtin_amdgcn_wmma_f32_16x16x32_bf16(false, a1, false, b3, (short)0, acc[1][3], false, false);
  };

  // pipeline prologue: stage tile 0
  issueB(0);
  loadA();
  storeA(0);
  asm volatile("s_wait_asynccnt 0" ::: "memory");
  __syncthreads();

  const int NK = K >> 5;
  for (int kt = 0; kt < NK; ++kt) {
    const int cur = kt & 1, nxt = cur ^ 1;
    const bool more = (kt + 1 < NK);
    if (more) {          // issue next tile's loads before this tile's math
      issueB(nxt);
      loadA();
      __builtin_prefetch(aptr, 0, 1);   // global_prefetch_b8 two tiles ahead
    }
    compute(cur);        // WMMAs overlap async copy + global A latency
    if (more) {
      storeA(nxt);       // waits loadcnt at first use of staged regs
      asm volatile("s_wait_asynccnt 0" ::: "memory");
    }
    __syncthreads();
  }

  const bool do_gelu = (act == ACT_GELU);
#pragma unroll
  for (int f = 0; f < 2; ++f)
#pragma unroll
    for (int g = 0; g < 4; ++g)
#pragma unroll
      for (int v = 0; v < 8; ++v) {
        int row = rowBase + wm * 32 + f * 16 + v + half * 8;
        int col = colBase + wn * 64 + g * 16 + l;
        float x = acc[f][g][v] + bias[col];
        if (do_gelu) x = gelu_exact(x);
        C[(size_t)row * N + col] = x;
      }
}

// ---------------------------------------------------------------------------
// Kernel 3: flash attention, WMMA for QK^T and P@V.
// Grid (SEQ/64, NHEAD, LLAY*BSZ), block 128 (4 waves), 64 q-rows per block.
// ---------------------------------------------------------------------------
__global__ __launch_bounds__(128)
void attn_kernel(const float* __restrict__ Q, const float* __restrict__ Kp,
                 const float* __restrict__ V, const float* __restrict__ scale,
                 float* __restrict__ O) {
  __shared__ __bf16 Qs[64][136];
  __shared__ __bf16 Ks[64][136];
  __shared__ __bf16 Vt[128][72];      // transposed V tile: [n][k2]
  __shared__ __bf16 Ps[4][16][72];    // per-wave P strip
  const int qt = blockIdx.x, h = blockIdx.y, lb = blockIdx.z;
  const int b = lb & 3;
  const int tid = threadIdx.x;
  const int wave = tid >> 5, lid = tid & 31, half = lid >> 4, l = lid & 15;
  const float qmul = scale[b * NHEAD + h] * 0.08838834764831845f; // 1/sqrt(128)
  const size_t base = (size_t)lb * SEQ * HDIM + (size_t)h * HD;

  {
    int d = (tid & 31) * 4, r0 = tid >> 5;
#pragma unroll
    for (int p = 0; p < 16; ++p) {
      int r = r0 + p * 4;
      float4 qv = *(const float4*)(Q + base + (size_t)(qt * 64 + r) * HDIM + d);
      unsigned* dst = (unsigned*)&Qs[r][d];
      dst[0] = pk2(qv.x * qmul, qv.y * qmul);
      dst[1] = pk2(qv.z * qmul, qv.w * qmul);
    }
  }

  v8f o[8];
  const v8f vz = {0.f,0.f,0.f,0.f,0.f,0.f,0.f,0.f};
#pragma unroll
  for (int n = 0; n < 8; ++n) o[n] = vz;
  float m_i[8], l_i[8];
#pragma unroll
  for (int v = 0; v < 8; ++v) { m_i[v] = -3.0e38f; l_i[v] = 0.f; }

  for (int kt = 0; kt < SEQ / 64; ++kt) {
    __syncthreads();
    {
      int d = (tid & 31) * 4, r0 = tid >> 5;
#pragma unroll
      for (int p = 0; p < 16; ++p) {
        int r = r0 + p * 4;
        size_t gofs = base + (size_t)(kt * 64 + r) * HDIM + d;
        float4 kv = *(const float4*)(Kp + gofs);
        unsigned* dst = (unsigned*)&Ks[r][d];
        dst[0] = pk2(kv.x, kv.y);
        dst[1] = pk2(kv.z, kv.w);
        float4 vv = *(const float4*)(V + gofs);
        Vt[d + 0][r] = f2bf(vv.x); Vt[d + 1][r] = f2bf(vv.y);
        Vt[d + 2][r] = f2bf(vv.z); Vt[d + 3][r] = f2bf(vv.w);
      }
    }
    __syncthreads();

    v8f s[4];
#pragma unroll
    for (int g = 0; g < 4; ++g) s[g] = vz;
#pragma unroll
    for (int ks = 0; ks < 128; ks += 32) {
      v16bf af = ld_frag(&Qs[wave * 16 + l][ks], half * 8);
#pragma unroll
      for (int g = 0; g < 4; ++g) {
        v16bf bf = ld_frag(&Ks[g * 16 + l][ks], half * 8);
        s[g] = __builtin_amdgcn_wmma_f32_16x16x32_bf16(false, af, false, bf,
                                                       (short)0, s[g], false, false);
      }
    }

#pragma unroll
    for (int v = 0; v < 8; ++v) {
      float mx = -3.0e38f;
#pragma unroll
      for (int g = 0; g < 4; ++g) mx = fmaxf(mx, s[g][v]);
      for (int msk = 1; msk < 16; msk <<= 1) mx = fmaxf(mx, __shfl_xor(mx, msk, 32));
      float mnew = fmaxf(m_i[v], mx);
      float alpha = __expf(m_i[v] - mnew);
      float ps = 0.f;
#pragma unroll
      for (int g = 0; g < 4; ++g) {
        float pv = __expf(s[g][v] - mnew);
        s[g][v] = pv; ps += pv;
      }
      for (int msk = 1; msk < 16; msk <<= 1) ps += __shfl_xor(ps, msk, 32);
      l_i[v] = l_i[v] * alpha + ps;
      m_i[v] = mnew;
#pragma unroll
      for (int n = 0; n < 8; ++n) o[n][v] *= alpha;
    }

#pragma unroll
    for (int v = 0; v < 8; ++v)
#pragma unroll
      for (int g = 0; g < 4; ++g)
        Ps[wave][v + half * 8][g * 16 + l] = f2bf(s[g][v]);

#pragma unroll
    for (int ks = 0; ks < 64; ks += 32) {
      v16bf af = ld_frag(&Ps[wave][l][ks], half * 8);
#pragma unroll
      for (int n = 0; n < 8; ++n) {
        v16bf bf = ld_frag(&Vt[n * 16 + l][ks], half * 8);
        o[n] = __builtin_amdgcn_wmma_f32_16x16x32_bf16(false, af, false, bf,
                                                       (short)0, o[n], false, false);
      }
    }
  }

#pragma unroll
  for (int v = 0; v < 8; ++v) {
    float inv = 1.f / l_i[v];
    int row = qt * 64 + wave * 16 + v + half * 8;
#pragma unroll
    for (int n = 0; n < 8; ++n)
      O[base + (size_t)row * HDIM + n * 16 + l] = o[n][v] * inv;
  }
}

// ---------------------------------------------------------------------------
// Kernel 4: layernorm (one block per 1024-wide row), optional residual add
// ---------------------------------------------------------------------------
__global__ __launch_bounds__(256)
void layernorm_kernel(const float* __restrict__ x, const float* __restrict__ res,
                      const float* __restrict__ g, const float* __restrict__ be,
                      float* __restrict__ out) {
  __shared__ float red[256];
  __shared__ float stats[2];
  const size_t row = blockIdx.x;
  const int t = threadIdx.x;
  float v[4];
  float s = 0.f;
#pragma unroll
  for (int i = 0; i < 4; ++i) {
    int c = t + i * 256;
    float val = x[row * HDIM + c];
    if (res) val += res[row * HDIM + c];
    v[i] = val; s += val;
  }
  red[t] = s; __syncthreads();
  for (int o = 128; o > 0; o >>= 1) { if (t < o) red[t] += red[t + o]; __syncthreads(); }
  if (t == 0) stats[0] = red[0] * (1.f / HDIM);
  __syncthreads();
  float mean = stats[0];
  float vs = 0.f;
#pragma unroll
  for (int i = 0; i < 4; ++i) { float d = v[i] - mean; vs += d * d; }
  red[t] = vs; __syncthreads();
  for (int o = 128; o > 0; o >>= 1) { if (t < o) red[t] += red[t + o]; __syncthreads(); }
  if (t == 0) stats[1] = red[0] * (1.f / HDIM);
  __syncthreads();
  float rstd = rsqrtf(stats[1] + 1e-5f);
#pragma unroll
  for (int i = 0; i < 4; ++i) {
    int c = t + i * 256;
    out[row * HDIM + c] = (v[i] - mean) * rstd * g[c] + be[c];
  }
}

// ---------------------------------------------------------------------------
// Kernel 5: routing-weighted combine + norm reduction; finalize; zero
// ---------------------------------------------------------------------------
__global__ void zero1_kernel(float* p) { if (threadIdx.x == 0) *p = 0.f; }

__global__ __launch_bounds__(256)
void combine_kernel(const float* __restrict__ fa, const float* __restrict__ me,
                    const float* __restrict__ fu, const float* __restrict__ probs,
                    float* __restrict__ out, float* __restrict__ norm_acc) {
  __shared__ float red[256];
  const size_t idx = (size_t)blockIdx.x * 256 + threadIdx.x;
  const int b = (int)((idx >> 20) & 3);    // idx / (SEQ*HDIM) mod BSZ
  float val = probs[b * 3 + 0] * fa[idx] + probs[b * 3 + 1] * me[idx] +
              probs[b * 3 + 2] * fu[idx];
  out[idx] = val;
  red[threadIdx.x] = val * val;
  __syncthreads();
  for (int o = 128; o > 0; o >>= 1) {
    if (threadIdx.x < o) red[threadIdx.x] += red[threadIdx.x + o];
    __syncthreads();
  }
  if (threadIdx.x == 0) atomicAdd(norm_acc, red[0]);
}

__global__ void finalize_kernel(const float* probs, const float* dec,
                                const float* norm_acc, float* out) {
  const int t = threadIdx.x;
  if (t < 12) out[NEL + t] = probs[t];
  if (t < 4)  out[NEL + 12 + t] = dec[t];
  if (t == 0) out[NEL + 16] = sqrtf(*norm_acc);
}

// ---------------------------------------------------------------------------
// Host orchestration
// ---------------------------------------------------------------------------
enum {
  IN_HS = 0, IN_UMAP,
  UE_W1, UE_B1, UE_W2, UE_B2, UE_W3, UE_B3,
  RT_W1, RT_B1, RT_W2, RT_B2,
  AT_WQ, AT_BQ, AT_WK, AT_BK, AT_WV, AT_BV, AT_WO, AT_BO,
  AT_US1, AT_UB1, AT_US2, AT_UB2, AT_G, AT_BE,
  FS_W, FS_B, FS_G, FS_BE,
  MD_W1, MD_B1, MD_W2, MD_B2, MD_G, MD_BE,
  FL_W1, FL_B1, FL_W2, FL_B2, FL_G, FL_BE
};

extern "C" void kernel_launch(void* const* d_in, const int* in_sizes, int n_in,
                              void* d_out, int out_size, void* d_ws, size_t ws_size,
                              hipStream_t stream) {
  (void)in_sizes; (void)n_in; (void)out_size; (void)ws_size;
  auto F = [&](int i) { return (const float*)d_in[i]; };
  const float* hs = F(IN_HS);                 // (LLAY*BSZ*SEQ, HDIM)
  float* out = (float*)d_out;
  float* ws  = (float*)d_ws;

  float* q       = ws;                        // also ping
  float* kbuf    = ws + NEL;                  // also pong
  float* vbuf    = ws + 2 * NEL;              // also fast_out
  float* att     = ws + 3 * NEL;              // also med_out
  float* enh     = ws + 4 * NEL;
  float* mid     = ws + 5 * NEL;              // 4*NEL (FFN 4H intermediate)
  float* fullout = ws + 9 * NEL;
  float* t2      = ws + 10 * NEL;
  float* smallp  = ws + 11 * NEL;
  float* uscale  = smallp;        // 32
  float* probs   = smallp + 32;   // 12
  float* dec     = smallp + 44;   // 4
  float* normacc = smallp + 48;   // 1
  __bf16* wt     = (__bf16*)(smallp + 64);    // 4M bf16 reusable weight buffer
  float* ping = q; float* pong = kbuf; float* fastout = vbuf; float* medout = att;

  auto convW = [&](const float* W, int K, int N) {
    convertT_kernel<<<dim3(N / 32, K / 32), 256, 0, stream>>>(W, wt, K, N);
  };

  // 1. uncertainty / routing / scale MLPs
  small_mlp_kernel<<<1, 256, 0, stream>>>(
      F(IN_UMAP), F(UE_W1), F(UE_B1), F(UE_W2), F(UE_B2), F(UE_W3), F(UE_B3),
      F(RT_W1), F(RT_B1), F(RT_W2), F(RT_B2),
      F(AT_US1), F(AT_UB1), F(AT_US2), F(AT_UB2),
      uscale, probs, dec);

  const dim3 gN1(HDIM / 128, MTOK / 128);     // N=1024
  const dim3 gN4(4 * HDIM / 128, MTOK / 128); // N=4096

  // 2. Q/K/V projections
  convW(F(AT_WQ), HDIM, HDIM);
  gemm_bias_act<<<gN1, 256, 0, stream>>>(hs, wt, F(AT_BQ), q,    MTOK, HDIM, HDIM, ACT_NONE);
  convW(F(AT_WK), HDIM, HDIM);
  gemm_bias_act<<<gN1, 256, 0, stream>>>(hs, wt, F(AT_BK), kbuf, MTOK, HDIM, HDIM, ACT_NONE);
  convW(F(AT_WV), HDIM, HDIM);
  gemm_bias_act<<<gN1, 256, 0, stream>>>(hs, wt, F(AT_BV), vbuf, MTOK, HDIM, HDIM, ACT_NONE);

  // 3. flash attention
  attn_kernel<<<dim3(SEQ / 64, NHEAD, LLAY * BSZ), 128, 0, stream>>>(q, kbuf, vbuf, uscale, att);

  // 4. output projection + residual layernorm -> enhanced
  convW(F(AT_WO), HDIM, HDIM);
  gemm_bias_act<<<gN1, 256, 0, stream>>>(att, wt, F(AT_BO), q, MTOK, HDIM, HDIM, ACT_NONE);
  layernorm_kernel<<<MTOK, 256, 0, stream>>>(q, hs, F(AT_G), F(AT_BE), enh);

  // 5. fast path
  {
    const float* cur = enh;
    for (int li = 0; li < 2; ++li) {
      convW(F(FS_W) + (size_t)li * HDIM * HDIM, HDIM, HDIM);
      gemm_bias_act<<<gN1, 256, 0, stream>>>(cur, wt, F(FS_B) + li * HDIM,
                                             mid, MTOK, HDIM, HDIM, ACT_GELU);
      float* dst = (li == 1) ? fastout : ping;
      layernorm_kernel<<<MTOK, 256, 0, stream>>>(mid, nullptr, F(FS_G) + li * HDIM,
                                                 F(FS_BE) + li * HDIM, dst);
      cur = dst;
    }
  }

  // 6. medium & full paths: 8x [LN(gelu(h@W1+b1)@W2+b2)]
  for (int path = 0; path < 2; ++path) {
    const float* W1 = F(path ? FL_W1 : MD_W1);
    const float* B1 = F(path ? FL_B1 : MD_B1);
    const float* W2 = F(path ? FL_W2 : MD_W2);
    const float* B2 = F(path ? FL_B2 : MD_B2);
    const float* G  = F(path ? FL_G  : MD_G);
    const float* BE = F(path ? FL_BE : MD_BE);
    float* final_dst = path ? fullout : medout;
    const float* cur = enh;
    for (int li = 0; li < 8; ++li) {
      convW(W1 + (size_t)li * HDIM * 4 * HDIM, HDIM, 4 * HDIM);
      gemm_bias_act<<<gN4, 256, 0, stream>>>(cur, wt, B1 + li * 4 * HDIM,
                                             mid, MTOK, 4 * HDIM, HDIM, ACT_GELU);
      convW(W2 + (size_t)li * 4 * HDIM * HDIM, 4 * HDIM, HDIM);
      gemm_bias_act<<<gN1, 256, 0, stream>>>(mid, wt, B2 + li * HDIM,
                                             t2, MTOK, HDIM, 4 * HDIM, ACT_NONE);
      float* dst = (li == 7) ? final_dst : ((li & 1) ? pong : ping);
      layernorm_kernel<<<MTOK, 256, 0, stream>>>(t2, nullptr, G + li * HDIM, BE + li * HDIM, dst);
      cur = dst;
    }
  }

  // 7. routed combine + outputs
  zero1_kernel<<<1, 32, 0, stream>>>(normacc);
  combine_kernel<<<(unsigned)(NEL / 256), 256, 0, stream>>>(fastout, medout, fullout,
                                                            probs, out, normacc);
  finalize_kernel<<<1, 32, 0, stream>>>(probs, dec, normacc, out);
}
